// AttentionHead_8735963480578
// MI455X (gfx1250) — matrix operand order repaired and checked by reading, not compiled, not run on hardware
//
#include <hip/hip_runtime.h>

typedef __bf16 bf16;
typedef __attribute__((ext_vector_type(16))) __bf16 v16bf;
typedef __attribute__((ext_vector_type(8)))  __bf16 v8bf;
typedef __attribute__((ext_vector_type(8)))  float  v8f;
typedef __attribute__((ext_vector_type(8)))  unsigned int v8u;
typedef __attribute__((ext_vector_type(4)))  float  v4f;

#define WMMA_BF16(a, b, c) \
  __builtin_amdgcn_wmma_f32_16x16x32_bf16(false, (a), false, (b), (short)0, (c), false, false)

static constexpr int BN = 4;     // batch
static constexpr int T  = 4096;  // sequence
static constexpr int D  = 1024;  // d_model
static constexpr int DK = 64;    // d_key

// ---------- fragment helpers (contiguous-k layouts, wave32) ----------

__device__ __forceinline__ v16bf cat8(v8bf a, v8bf b) {
  v16bf r;
#pragma unroll
  for (int i = 0; i < 8; ++i) { r[i] = a[i]; r[i + 8] = b[i]; }
  return r;
}

// A-matrix 16x32 bf16 from row-major source (row stride ld elements).
// lanes 0-15: row=lane, K=0..15; lanes 16-31: row=lane-16, K=16..31
__device__ __forceinline__ v16bf afrag(const bf16* base, int ld, int lane) {
  int lr = lane & 15, h = lane >> 4;
  const bf16* p = base + lr * ld + h * 16;
  return cat8(*(const v8bf*)p, *(const v8bf*)(p + 8));
}

// B-matrix 32x16 bf16 where B[k][n] = src[n*ld + k].
// lanes 0-15: col=lane, K=0..15; lanes 16-31: col=lane-16, K=16..31
__device__ __forceinline__ v16bf bfrag(const bf16* base, int ld, int lane) {
  int n = lane & 15, h = lane >> 4;
  const bf16* p = base + n * ld + h * 16;
  return cat8(*(const v8bf*)p, *(const v8bf*)(p + 8));
}

// ---------- kernel 1: pack Wq/Wk/Wv into WMMA-B bf16 fragment order ----------
// wpack[((w*32 + kt)*4 + nt)*256 + lane*8 + j] = pack(W[kpb][n], W[kpb+1][n])
__global__ void pack_w_kernel(const float* __restrict__ Wq, const float* __restrict__ Wk,
                              const float* __restrict__ Wv, unsigned int* __restrict__ wpack) {
  int tid = blockIdx.x * blockDim.x + threadIdx.x;
  if (tid >= 3 * 32 * 4 * 32 * 8) return;
  int j  = tid & 7;
  int l  = (tid >> 3) & 31;
  int nt = (tid >> 8) & 3;
  int kt = (tid >> 10) & 31;
  int w  = tid >> 15;
  const float* W = (w == 0) ? Wq : ((w == 1) ? Wk : Wv);
  int n   = nt * 16 + (l & 15);
  int kpb = kt * 32 + (l & 16) + 2 * j;
  bf16 lo = (bf16)W[(size_t)kpb * DK + n];
  bf16 hi = (bf16)W[(size_t)(kpb + 1) * DK + n];
  unsigned int ulo = __builtin_bit_cast(unsigned short, lo);
  unsigned int uhi = __builtin_bit_cast(unsigned short, hi);
  wpack[tid] = ulo | (uhi << 16);
}

// ---------- kernel 2: q/k/v projections via WMMA bf16 ----------
// grid 256 blocks, 128 threads (4 waves). Block: 64 rows of x @ all 3 weights.
__global__ __launch_bounds__(128) void proj_qkv_kernel(
    const float* __restrict__ x, const unsigned int* __restrict__ wpack,
    bf16* __restrict__ qo, bf16* __restrict__ ko, bf16* __restrict__ vo) {
  constexpr int LDP = 40;  // padded f32 row stride in LDS
  __shared__ __align__(16) float xs[64 * LDP];

  int tid = threadIdx.x;
  int lane = tid & 31, wv = tid >> 5;
  int lr = lane & 15, h = lane >> 4;
  int rb = blockIdx.x * 64;

  v8f acc[12];
#pragma unroll
  for (int t = 0; t < 12; ++t)
#pragma unroll
    for (int i = 0; i < 8; ++i) acc[t][i] = 0.0f;

  for (int kt = 0; kt < 32; ++kt) {
    __syncthreads();
    // stage x tile [64 rows][32 cols] f32 (coalesced b128 loads)
#pragma unroll
    for (int i = 0; i < 4; ++i) {
      int f = tid * 4 + i;        // float4 index 0..511
      int row = f >> 3, c4 = f & 7;
      v4f val = *(const v4f*)(x + (size_t)(rb + row) * D + kt * 32 + c4 * 4);
      *(v4f*)(&xs[row * LDP + c4 * 4]) = val;
    }
    __syncthreads();

    // build bf16 A-frag from LDS (convert f32 -> bf16)
    const float* pr = &xs[(wv * 16 + lr) * LDP];
    v16bf a;
#pragma unroll
    for (int e = 0; e < 8; ++e) a[e] = (bf16)pr[h * 16 + e];
#pragma unroll
    for (int e = 0; e < 8; ++e) a[e + 8] = (bf16)pr[h * 16 + 8 + e];

#pragma unroll
    for (int t = 0; t < 12; ++t) {
      const unsigned int* wp =
          wpack + ((size_t)(((t >> 2) * 32 + kt) * 4 + (t & 3)) * 32 + lane) * 8;
      v16bf bf = __builtin_bit_cast(v16bf, *(const v8u*)wp);
      acc[t] = WMMA_BF16(a, bf, acc[t]);
    }
  }

  // store C tiles (row = r + 8*h, col = lane&15 within 16-wide tile) as bf16 row-major
#pragma unroll
  for (int t = 0; t < 12; ++t) {
    bf16* out = (t < 4) ? qo : ((t < 8) ? ko : vo);
    int nt = t & 3;
#pragma unroll
    for (int r = 0; r < 8; ++r) {
      int row = rb + wv * 16 + r + 8 * h;
      out[(size_t)row * DK + nt * 16 + lr] = (bf16)acc[t][r];
    }
  }
}

// ---------- kernel 3: flash attention (causal) ----------
// grid (T/64, B), 128 threads (4 waves). Wave handles 16 query rows.
__global__ __launch_bounds__(128) void attn_kernel(
    const bf16* __restrict__ qg, const bf16* __restrict__ kg,
    const bf16* __restrict__ vg, float* __restrict__ out) {
  __shared__ __align__(16) bf16 Vt[64 * 32];       // V^T tile: [dk][key]
  __shared__ __align__(16) bf16 Pl[4][16 * 32];    // per-wave P staging

  int tid = threadIdx.x, lane = tid & 31, wv = tid >> 5;
  int lr = lane & 15, h = lane >> 4;
  int b = blockIdx.y;
  int qb64 = blockIdx.x * 64;
  const bf16* q = qg + (size_t)b * T * DK;
  const bf16* k = kg + (size_t)b * T * DK;
  const bf16* v = vg + (size_t)b * T * DK;
  int qwb = qb64 + wv * 16;

  // Q fragments (held for whole loop)
  v16bf qf0 = afrag(q + (size_t)qwb * DK + 0,  DK, lane);
  v16bf qf1 = afrag(q + (size_t)qwb * DK + 32, DK, lane);

  v8f o[4];
  float m[8], lsum[8];
#pragma unroll
  for (int nt = 0; nt < 4; ++nt)
#pragma unroll
    for (int i = 0; i < 8; ++i) o[nt][i] = 0.0f;
#pragma unroll
  for (int r = 0; r < 8; ++r) { m[r] = -3.0e38f; lsum[r] = 0.0f; }

  const float sc = (1.0f / 32.0f) * 1.4426950408889634f;  // 1/sqrt(D) * log2(e)
  int niter = qb64 / 32 + 2;
  int vrow = tid & 31, vcb = (tid >> 5) * 16;

  for (int it = 0; it < niter; ++it) {
    int jt = it * 32;

    // stage V^T tile into LDS (coalesced global b128, scattered b16 LDS stores)
    v8bf c0 = *(const v8bf*)(v + (size_t)(jt + vrow) * DK + vcb);
    v8bf c1 = *(const v8bf*)(v + (size_t)(jt + vrow) * DK + vcb + 8);
#pragma unroll
    for (int i = 0; i < 8; ++i) {
      Vt[(vcb + i) * 32 + vrow]     = c0[i];
      Vt[(vcb + 8 + i) * 32 + vrow] = c1[i];
    }
    __syncthreads();

    if (it + 1 < niter) {
      __builtin_prefetch(k + (size_t)(jt + 32) * DK, 0, 0);
      __builtin_prefetch(v + (size_t)(jt + 32) * DK, 0, 0);
    }

    // S = Q @ K^T for two 16-wide key subtiles (k row-major == B-frag layout)
    v8f s0, s1;
    {
      v8f z;
#pragma unroll
      for (int i = 0; i < 8; ++i) z[i] = 0.0f;
      v16bf b00 = bfrag(k + (size_t)(jt) * DK + 0,  DK, lane);
      v16bf b01 = bfrag(k + (size_t)(jt) * DK + 32, DK, lane);
      s0 = WMMA_BF16(qf0, b00, z);
      s0 = WMMA_BF16(qf1, b01, s0);
      v16bf b10 = bfrag(k + (size_t)(jt + 16) * DK + 0,  DK, lane);
      v16bf b11 = bfrag(k + (size_t)(jt + 16) * DK + 32, DK, lane);
      s1 = WMMA_BF16(qf0, b10, z);
      s1 = WMMA_BF16(qf1, b11, s1);
    }

    // scale (log2-domain) + causal mask; element r <-> (row qwb+r+8h, key jt+{lr, 16+lr})
    float p0[8], p1[8], tm[8];
#pragma unroll
    for (int r = 0; r < 8; ++r) {
      int qi = qwb + r + 8 * h;
      p0[r] = (jt + lr      <= qi) ? s0[r] * sc : -1.0e30f;
      p1[r] = (jt + 16 + lr <= qi) ? s1[r] * sc : -1.0e30f;
      tm[r] = fmaxf(p0[r], p1[r]);
    }
    // row max across the 16 lanes of each half
#pragma unroll
    for (int mk = 1; mk < 16; mk <<= 1)
#pragma unroll
      for (int r = 0; r < 8; ++r) tm[r] = fmaxf(tm[r], __shfl_xor(tm[r], mk, 32));

    float rs[8];
#pragma unroll
    for (int r = 0; r < 8; ++r) {
      float mn = fmaxf(m[r], tm[r]);
      float al = __builtin_amdgcn_exp2f(m[r] - mn);
      m[r] = mn;
      p0[r] = __builtin_amdgcn_exp2f(p0[r] - mn);
      p1[r] = __builtin_amdgcn_exp2f(p1[r] - mn);
      rs[r] = p0[r] + p1[r];
      lsum[r] *= al;
#pragma unroll
      for (int nt = 0; nt < 4; ++nt) o[nt][r] *= al;
    }
#pragma unroll
    for (int mk = 1; mk < 16; mk <<= 1)
#pragma unroll
      for (int r = 0; r < 8; ++r) rs[r] += __shfl_xor(rs[r], mk, 32);
#pragma unroll
    for (int r = 0; r < 8; ++r) lsum[r] += rs[r];

    // C-layout P -> row-major bf16 in per-wave LDS (same-wave LDS is in-order)
    bf16* pl = &Pl[wv][0];
#pragma unroll
    for (int r = 0; r < 8; ++r) {
      pl[(r + 8 * h) * 32 + lr]      = (bf16)p0[r];
      pl[(r + 8 * h) * 32 + 16 + lr] = (bf16)p1[r];
    }
    // re-read P as A-frag
    v16bf pf;
    {
      const bf16* pp = pl + lr * 32 + h * 16;
      pf = cat8(*(const v8bf*)pp, *(const v8bf*)(pp + 8));
    }
    // O += P @ V  (V^T rows in LDS give contiguous B-frags)
#pragma unroll
    for (int nt = 0; nt < 4; ++nt) {
      const bf16* vp = &Vt[(nt * 16 + lr) * 32 + h * 16];
      v16bf vf = cat8(*(const v8bf*)vp, *(const v8bf*)(vp + 8));
      o[nt] = WMMA_BF16(pf, vf, o[nt]);
    }
    __syncthreads();  // protect Vt before next stage
  }

  // normalize and store f32 output
#pragma unroll
  for (int r = 0; r < 8; ++r) {
    float inv = 1.0f / lsum[r];
    int row = qwb + r + 8 * h;
#pragma unroll
    for (int nt = 0; nt < 4; ++nt)
      out[((size_t)b * T + row) * DK + nt * 16 + lr] = o[nt][r] * inv;
  }
}

// ---------- launcher ----------
extern "C" void kernel_launch(void* const* d_in, const int* in_sizes, int n_in,
                              void* d_out, int out_size, void* d_ws, size_t ws_size,
                              hipStream_t stream) {
  const float* x  = (const float*)d_in[0];
  const float* Wq = (const float*)d_in[1];
  const float* Wk = (const float*)d_in[2];
  const float* Wv = (const float*)d_in[3];
  float* out = (float*)d_out;

  char* ws = (char*)d_ws;
  bf16* qb = (bf16*)(ws);
  bf16* kb = (bf16*)(ws + (size_t)2 * 1024 * 1024);
  bf16* vb = (bf16*)(ws + (size_t)4 * 1024 * 1024);
  unsigned int* wpack = (unsigned int*)(ws + (size_t)6 * 1024 * 1024);

  pack_w_kernel<<<384, 256, 0, stream>>>(Wq, Wk, Wv, wpack);
  proj_qkv_kernel<<<(BN * T) / 64, 128, 0, stream>>>(x, wpack, qb, kb, vb);
  attn_kernel<<<dim3(T / 64, BN), 128, 0, stream>>>(qb, kb, vb, out);
}
